// DLRM_Net_26474178413218
// MI455X (gfx1250) — matrix operand order repaired and checked by reading, not compiled, not run on hardware
//
#include <hip/hip_runtime.h>

typedef __attribute__((ext_vector_type(16))) _Float16 v16h;
typedef __attribute__((ext_vector_type(8)))  _Float16 v8h;
typedef __attribute__((ext_vector_type(8)))  float    v8f;
typedef __attribute__((ext_vector_type(4)))  int      v4i;

#define BATCH   16384
#define NTBL    26
#define NROWSZ  100000
#define EMBD    128
#define BSTRIDE 40      // padded f16 row stride for LDS B-tile (bank-conflict-free)
#define TSTRIDE 136     // padded f16 row stride for interaction T tile

// CDNA5 async global->LDS path (guarded; falls back to sync copy)
#if defined(__gfx1250__) && __has_builtin(__builtin_amdgcn_global_load_async_to_lds_b128) && __has_builtin(__builtin_amdgcn_s_wait_asynccnt)
#define HAVE_ASYNC_LDS 1
#else
#define HAVE_ASYNC_LDS 0
#endif

__device__ __forceinline__ v8f wmma16x16x32(v16h a, v16h b, v8f c) {
  return __builtin_amdgcn_wmma_f32_16x16x32_f16(
      /*neg_a=*/false, a, /*neg_b=*/false, b,
      /*c_mod=*/(short)0, c, /*reuse_a=*/false, /*reuse_b=*/false);
}

// copy 16 bytes global -> LDS (async when available)
__device__ __forceinline__ void cp16_g2l(_Float16* dst, const _Float16* src) {
#if HAVE_ASYNC_LDS
  __builtin_amdgcn_global_load_async_to_lds_b128(
      (__attribute__((address_space(1))) v4i*)src,
      (__attribute__((address_space(3))) v4i*)dst,
      /*offset=*/0, /*cpol=*/0);
#else
  *(uint4*)dst = *(const uint4*)src;
#endif
}

template <int N>
__device__ __forceinline__ void async_wait_le() {
#if HAVE_ASYNC_LDS
  __builtin_amdgcn_s_wait_asynccnt(N);
#endif
}

__device__ __forceinline__ v16h join16(v8h lo, v8h hi) {
  v16h r;
#pragma unroll
  for (int e = 0; e < 8; ++e) { r[e] = lo[e]; r[8 + e] = hi[e]; }
  return r;
}

// ---------------------------------------------------------------------------
// Convert f32 weight (N,K) row-major -> f16 (N,Kp) row-major, zero-padded K.
// ---------------------------------------------------------------------------
__global__ void cvt_weight(const float* __restrict__ W, _Float16* __restrict__ Wh,
                           int N, int K, int Kp) {
  int i = blockIdx.x * blockDim.x + threadIdx.x;
  if (i >= N * Kp) return;
  int n = i / Kp, k = i % Kp;
  Wh[i] = (k < K) ? (_Float16)W[(size_t)n * K + k] : (_Float16)0.0f;
}

// Pad dense_x (B,13) f32 -> (B,32) f16 with zeros.
__global__ void pad_dense_h(const float* __restrict__ x, _Float16* __restrict__ xh) {
  int i = blockIdx.x * blockDim.x + threadIdx.x;   // B*32 threads
  int b = i >> 5, c = i & 31;
  xh[i] = (c < 13) ? (_Float16)x[(size_t)b * 13 + c] : (_Float16)0.0f;
}

// ---------------------------------------------------------------------------
// Fused GEMM: Oh[M,N] = f16(act(Ah[M,Kp] @ Wh[N,Kp]^T + bias)), M = BATCH.
// Block = 256 threads = 8 waves. Block tile 128M x 128N; wave tile 32M x 64N
// (2 A-fragments x 4 B-fragments = 8 WMMAs per 32-wide k-step).
// B-tile (128N x 32K f16 = 8KB) staged in LDS, double-buffered, streamed with
// async global->LDS copies. grid = (N/128, BATCH/128).
// ---------------------------------------------------------------------------
__global__ void __launch_bounds__(256)
gemm_bias_act(const _Float16* __restrict__ Ah, const _Float16* __restrict__ Wh,
              const float* __restrict__ bias, _Float16* __restrict__ Oh,
              int Kp, int N, int do_relu) {
  __shared__ _Float16 Bsh[2][128][BSTRIDE];   // 20 KB

  const int lane = threadIdx.x & 31;
  const int wave = threadIdx.x >> 5;
  const int l15  = lane & 15;
  const int hi   = lane >> 4;
  const int mw   = wave & 3;        // 4 waves along M
  const int nw   = wave >> 2;       // 2 waves along N
  const int m0   = blockIdx.y * 128 + mw * 32;
  const int n0   = blockIdx.x * 128;
  const int nsub = nw * 64;

  // cooperative B-tile copy mapping: thread -> (row, 16-f16 half)
  const int crow  = threadIdx.x >> 1;
  const int chalf = (threadIdx.x & 1) * 16;
  const _Float16* wsrc_base = Wh + (size_t)(n0 + crow) * Kp + chalf;

  const _Float16* arow0 = Ah + (size_t)(m0 + l15) * Kp;
  const _Float16* arow1 = Ah + (size_t)(m0 + 16 + l15) * Kp;

  v8f acc[2][4];
#pragma unroll
  for (int p = 0; p < 2; ++p)
#pragma unroll
    for (int j = 0; j < 4; ++j) acc[p][j] = (v8f){};

  // prologue: stage tile 0 (and tile 1) into LDS
  {
    const _Float16* s = wsrc_base;
    cp16_g2l(&Bsh[0][crow][chalf], s);
    cp16_g2l(&Bsh[0][crow][chalf + 8], s + 8);
    if (Kp > 32) {
      cp16_g2l(&Bsh[1][crow][chalf], s + 32);
      cp16_g2l(&Bsh[1][crow][chalf + 8], s + 40);
    }
  }

  int cur = 0;
  for (int k0 = 0; k0 < Kp; k0 += 32) {
    if (k0 + 32 < Kp) async_wait_le<2>(); else async_wait_le<0>();
    __syncthreads();

    // A fragments (f16 direct, two 16B chunks per row)
    v16h a0 = join16(*(const v8h*)(arow0 + k0 + 8 * hi),
                     *(const v8h*)(arow0 + k0 + 16 + 8 * hi));
    v16h a1 = join16(*(const v8h*)(arow1 + k0 + 8 * hi),
                     *(const v8h*)(arow1 + k0 + 16 + 8 * hi));

#pragma unroll
    for (int j = 0; j < 4; ++j) {
      const _Float16* br = &Bsh[cur][nsub + j * 16 + l15][8 * hi];
      v16h b = join16(*(const v8h*)br, *(const v8h*)(br + 16));
      acc[0][j] = wmma16x16x32(a0, b, acc[0][j]);
      acc[1][j] = wmma16x16x32(a1, b, acc[1][j]);
    }
    __syncthreads();

    if (k0 + 64 < Kp) {
      const _Float16* s = wsrc_base + k0 + 64;
      cp16_g2l(&Bsh[cur][crow][chalf], s);
      cp16_g2l(&Bsh[cur][crow][chalf + 8], s + 8);
    }
    cur ^= 1;
  }

  // epilogue: C/D layout lane n = lane&15, vgpr v -> m = v + 8*hi
#pragma unroll
  for (int p = 0; p < 2; ++p) {
#pragma unroll
    for (int j = 0; j < 4; ++j) {
      int n = n0 + nsub + j * 16 + l15;
      float bv = bias[n];
#pragma unroll
      for (int v = 0; v < 8; ++v) {
        int m = m0 + p * 16 + v + 8 * hi;
        float val = acc[p][j][v] + bv;
        if (do_relu) val = fmaxf(val, 0.0f);
        Oh[(size_t)m * N + n] = (_Float16)val;
      }
    }
  }
}

// ---------------------------------------------------------------------------
// Interaction: one wave per sample. Gather T = [x[b]; 26 emb rows] (27x128)
// into LDS f16 (padded to 32 rows, stride 136), compute lower-tri of Z = T T^T
// with WMMA, emit z[b] = concat(x[b], tril(Z,k=-1)) f16, padded to 512 cols.
// Block = 128 threads = 4 waves = 4 samples. grid = BATCH/4.
// ---------------------------------------------------------------------------
__global__ void __launch_bounds__(128)
interact(const _Float16* __restrict__ Xh, const float* __restrict__ emb,
         const int* __restrict__ lSi, _Float16* __restrict__ Z) {
  __shared__ _Float16 Tsh[4][32][TSTRIDE];   // 34 KB
  const int lane = threadIdx.x & 31;
  const int wave = threadIdx.x >> 5;
  const int b = blockIdx.x * 4 + wave;
  _Float16* Tw = &Tsh[wave][0][0];

  // row 0: bottom-MLP output x[b] (already f16): 4 f16 per lane
  *(uint2*)(Tw + lane * 4) = *(const uint2*)(Xh + (size_t)b * 128 + lane * 4);

  // rows 1..26: gathered embedding rows (bag size is exactly 1), f32 -> f16
#pragma unroll 2
  for (int t = 0; t < NTBL; ++t) {
    int idx = lSi[(size_t)t * BATCH + b];
    float4 f = ((const float4*)(emb + ((size_t)t * NROWSZ + idx) * EMBD))[lane];
    _Float16* d = Tw + (t + 1) * TSTRIDE + lane * 4;
    d[0] = (_Float16)f.x; d[1] = (_Float16)f.y; d[2] = (_Float16)f.z; d[3] = (_Float16)f.w;
  }
  // rows 27..31: zero padding
#pragma unroll
  for (int r = 27; r < 32; ++r) {
    _Float16* d = Tw + r * TSTRIDE + lane * 4;
    d[0] = (_Float16)0.0f; d[1] = (_Float16)0.0f; d[2] = (_Float16)0.0f; d[3] = (_Float16)0.0f;
  }
  __syncthreads();

  const int l15 = lane & 15;
  const int hi  = lane >> 4;

  v8f acc0 = (v8f){}, acc1 = (v8f){}, acc2 = (v8f){};
#pragma unroll
  for (int k0 = 0; k0 < 128; k0 += 32) {
    const _Float16* pa0 = Tw + (0 * 16 + l15) * TSTRIDE + k0 + 8 * hi;
    const _Float16* pa1 = Tw + (1 * 16 + l15) * TSTRIDE + k0 + 8 * hi;
    v16h a0 = join16(*(const v8h*)pa0, *(const v8h*)(pa0 + 16));
    v16h a1 = join16(*(const v8h*)pa1, *(const v8h*)(pa1 + 16));

    const _Float16* pb0 = Tw + (0 * 16 + l15) * TSTRIDE + k0 + 8 * hi;
    const _Float16* pb1 = Tw + (1 * 16 + l15) * TSTRIDE + k0 + 8 * hi;
    v16h b0 = join16(*(const v8h*)pb0, *(const v8h*)(pb0 + 16));
    v16h b1 = join16(*(const v8h*)pb1, *(const v8h*)(pb1 + 16));

    acc0 = wmma16x16x32(a0, b0, acc0);   // tile (0,0)
    acc1 = wmma16x16x32(a1, b0, acc1);   // tile (1,0)
    acc2 = wmma16x16x32(a1, b1, acc2);   // tile (1,1)
  }

  _Float16* zrow = Z + (size_t)b * 512;
  // z[0..127] = x[b]
  *(uint2*)(zrow + lane * 4) = *(const uint2*)(Xh + (size_t)b * 128 + lane * 4);
  // z[479..511] = 0
  zrow[479 + lane] = (_Float16)0.0f;
  if (lane == 0) zrow[511] = (_Float16)0.0f;

  // tril(Z, k=-1) row-major: (i,j), j<i  ->  i*(i-1)/2 + j
  const int j0 = l15;
#pragma unroll
  for (int v = 0; v < 8; ++v) {
    { int i = v + 8 * hi,      j = j0;      if (i < 27 && j < i) zrow[128 + (i * (i - 1)) / 2 + j] = (_Float16)acc0[v]; }
    { int i = 16 + v + 8 * hi, j = j0;      if (i < 27 && j < i) zrow[128 + (i * (i - 1)) / 2 + j] = (_Float16)acc1[v]; }
    { int i = 16 + v + 8 * hi, j = 16 + j0; if (i < 27 && j < i) zrow[128 + (i * (i - 1)) / 2 + j] = (_Float16)acc2[v]; }
  }
}

// ---------------------------------------------------------------------------
// Final layer: out[b] = sigmoid(dot(a[b,0:256], w) + bias)  (N=1, VALU)
// ---------------------------------------------------------------------------
__global__ void __launch_bounds__(256)
final_dot_sigmoid(const _Float16* __restrict__ A, const float* __restrict__ w,
                  const float* __restrict__ bias, float* __restrict__ out) {
  int b = blockIdx.x * blockDim.x + threadIdx.x;
  const _Float16* a = A + (size_t)b * 256;
  float s = bias[0];
#pragma unroll 8
  for (int k = 0; k < 256; ++k) s = fmaf((float)a[k], w[k], s);
  out[b] = 1.0f / (1.0f + expf(-s));
}

// ---------------------------------------------------------------------------
extern "C" void kernel_launch(void* const* d_in, const int* in_sizes, int n_in,
                              void* d_out, int out_size, void* d_ws, size_t ws_size,
                              hipStream_t stream) {
  const float* dense_x = (const float*)d_in[0];
  const int*   lSi     = (const int*)d_in[2];          // (26, B)
  const float* emb     = (const float*)d_in[3];
  const float* bw0 = (const float*)d_in[4],  *bb0 = (const float*)d_in[5];
  const float* bw1 = (const float*)d_in[6],  *bb1 = (const float*)d_in[7];
  const float* bw2 = (const float*)d_in[8],  *bb2 = (const float*)d_in[9];
  const float* tw0 = (const float*)d_in[10], *tb0 = (const float*)d_in[11];
  const float* tw1 = (const float*)d_in[12], *tb1 = (const float*)d_in[13];
  const float* tw2 = (const float*)d_in[14], *tb2 = (const float*)d_in[15];
  const float* tw3 = (const float*)d_in[16], *tb3 = (const float*)d_in[17];
  const float* tw4 = (const float*)d_in[18], *tb4 = (const float*)d_in[19];
  float* out = (float*)d_out;

  // ---- workspace layout: f16 activations, then f16 weights ----
  _Float16* ws = (_Float16*)d_ws;
  size_t off = 0;
  _Float16* xh   = ws + off; off += (size_t)BATCH * 32;    // padded dense input
  _Float16* a512 = ws + off; off += (size_t)BATCH * 512;   // 512-wide scratch
  _Float16* a256 = ws + off; off += (size_t)BATCH * 256;   // 256-wide scratch
  _Float16* x128 = ws + off; off += (size_t)BATCH * 128;   // bottom MLP output
  _Float16* z512 = ws + off; off += (size_t)BATCH * 512;   // interaction out (479 -> 512)
  _Float16* u0   = ws + off; off += (size_t)BATCH * 1024;
  _Float16* u1   = ws + off; off += (size_t)BATCH * 1024;

  _Float16* bw0h = ws + off; off += (size_t)512 * 32;
  _Float16* bw1h = ws + off; off += (size_t)256 * 512;
  _Float16* bw2h = ws + off; off += (size_t)128 * 256;
  _Float16* tw0h = ws + off; off += (size_t)1024 * 512;
  _Float16* tw1h = ws + off; off += (size_t)1024 * 1024;
  _Float16* tw2h = ws + off; off += (size_t)512 * 1024;
  _Float16* tw3h = ws + off; off += (size_t)256 * 512;

  auto cvt = [&](const float* W, _Float16* Wh, int N, int K, int Kp) {
    int total = N * Kp;
    cvt_weight<<<(total + 255) / 256, 256, 0, stream>>>(W, Wh, N, K, Kp);
  };
  cvt(bw0, bw0h, 512, 13, 32);
  cvt(bw1, bw1h, 256, 512, 512);
  cvt(bw2, bw2h, 128, 256, 256);
  cvt(tw0, tw0h, 1024, 479, 512);
  cvt(tw1, tw1h, 1024, 1024, 1024);
  cvt(tw2, tw2h, 512, 1024, 1024);
  cvt(tw3, tw3h, 256, 512, 512);

  pad_dense_h<<<(BATCH * 32 + 255) / 256, 256, 0, stream>>>(dense_x, xh);

  auto gemm = [&](const _Float16* A, const _Float16* Wh, const float* bias,
                  _Float16* O, int Kp, int N, int relu) {
    dim3 grid(N / 128, BATCH / 128);
    gemm_bias_act<<<grid, 256, 0, stream>>>(A, Wh, bias, O, Kp, N, relu);
  };

  // Bottom MLP: 13(->32) -> 512 -> 256 -> 128
  gemm(xh,   bw0h, bb0, a512, 32, 512, 1);
  gemm(a512, bw1h, bb1, a256, 512, 256, 1);
  gemm(a256, bw2h, bb2, x128, 256, 128, 1);

  // Gather + pairwise interaction (Z = T T^T lower triangle) -> z (B,512)
  interact<<<BATCH / 4, 128, 0, stream>>>(x128, emb, lSi, z512);

  // Top MLP: 479(->512) -> 1024 -> 1024 -> 512 -> 256 -> 1
  gemm(z512, tw0h, tb0, u0, 512, 1024, 1);
  gemm(u0,   tw1h, tb1, u1, 1024, 1024, 1);
  gemm(u1,   tw2h, tb2, a512, 1024, 512, 1);
  gemm(a512, tw3h, tb3, a256, 512, 256, 1);
  final_dot_sigmoid<<<BATCH / 256, 256, 0, stream>>>(a256, tw4, tb4, out);
}
